// PrimalTRMCore_50663434223825
// MI455X (gfx1250) — matrix operand order repaired and checked by reading, not compiled, not run on hardware
//
#include <hip/hip_runtime.h>
#include <stdint.h>

typedef __attribute__((ext_vector_type(16))) __bf16 v16bf;
typedef __attribute__((ext_vector_type(8)))  float  v8f;

#define M_DIM 8192
#define N_DIM 4096
#define K_DIM 4096
#define LDSTR 72          // LDS row stride (bf16 elems): 144 B, 16B-aligned, bank-spreading
#define KTILE 64          // K staged per buffer (2 WMMA sub-steps of 32)
#define NT2   (K_DIM / KTILE)

#if __has_builtin(__builtin_amdgcn_s_wait_asynccnt)
#define WAIT_ASYNC(n) __builtin_amdgcn_s_wait_asynccnt(n)
#else
#define WAIT_ASYNC(n) asm volatile("s_wait_asynccnt %0" :: "i"(n) : "memory")
#endif

// CDNA5 async copy: global -> LDS directly, tracked by ASYNCcnt (ISA 15.18.3)
__device__ __forceinline__ void async_cp_b128(unsigned lds_off, const unsigned short* g) {
    asm volatile("global_load_async_to_lds_b128 %0, %1, off"
                 :: "v"(lds_off), "v"(g)
                 : "memory");
}

__device__ __forceinline__ unsigned short f32_to_bf16(float f) {
    union { float f; uint32_t u; } c; c.f = f;
    uint32_t u = c.u;
    u += 0x7fffu + ((u >> 16) & 1u);   // round-to-nearest-even
    return (unsigned short)(u >> 16);
}

// ---- pass 1: x f32 -> bf16 ----------------------------------------------
__global__ __launch_bounds__(256) void cvt_x(const float* __restrict__ x,
                                             unsigned short* __restrict__ xb) {
    const size_t i = ((size_t)blockIdx.x * blockDim.x + threadIdx.x) * 4;
    float4 v = *(const float4*)(x + i);
    ushort4 r;
    r.x = f32_to_bf16(v.x); r.y = f32_to_bf16(v.y);
    r.z = f32_to_bf16(v.z); r.w = f32_to_bf16(v.w);
    *(ushort4*)(xb + i) = r;
}

// ---- pass 2: LUT dequant -> bf16 W[o][i] (row-major, K contiguous) ------
__global__ __launch_bounds__(256) void dequant_w(const int* __restrict__ base_idx,
                                                 const int* __restrict__ fine_idx,
                                                 const float* __restrict__ scale,
                                                 const float* __restrict__ lut,
                                                 unsigned short* __restrict__ wb) {
    const size_t i = ((size_t)blockIdx.x * blockDim.x + threadIdx.x) * 4;
    const int o = (int)(i >> 12);          // / 4096
    const float s = scale[o];
    int4 b = *(const int4*)(base_idx + i);
    int4 f = *(const int4*)(fine_idx + i);
    ushort4 r;
    r.x = f32_to_bf16(lut[b.x * 256 + f.x] * s);
    r.y = f32_to_bf16(lut[b.y * 256 + f.y] * s);
    r.z = f32_to_bf16(lut[b.z * 256 + f.z] * s);
    r.w = f32_to_bf16(lut[b.w * 256 + f.w] * s);
    *(ushort4*)(wb + i) = r;
}

// ---- pass 3: bf16 WMMA GEMM, async double-buffered LDS pipeline ----------
// out[M][N] = Xb[M][K] * Wb[N][K]^T ; block 128x128, wave tile 32x64, K staged 64
__global__ __launch_bounds__(256) void gemm_bf16(const unsigned short* __restrict__ Xb,
                                                 const unsigned short* __restrict__ Wb,
                                                 float* __restrict__ out) {
    __shared__ __align__(16) unsigned short As[2][128 * LDSTR]; // 2 x (128x64) tiles
    __shared__ __align__(16) unsigned short Bs[2][128 * LDSTR];

    const int tid   = threadIdx.x;
    const int lane  = tid & 31;
    const int wave  = tid >> 5;      // 0..7
    const int waveM = wave & 3;      // 4 waves along M: 32 rows each
    const int waveN = wave >> 2;     // 2 waves along N: 64 cols each
    const int m0 = blockIdx.y * 128;
    const int n0 = blockIdx.x * 128;

    // cooperative-load mapping: 128 rows x 2 segs of 32 bf16 (64 B) per tile
    const int row = tid >> 1;        // 0..127
    const int seg = tid & 1;         // 0..1

    const int lh = lane >> 4;        // lane half (0/1)
    const int ll = lane & 15;

    // per-thread global bases (element offsets)
    const unsigned short* gA = Xb + (size_t)(m0 + row) * K_DIM + seg * 32;
    const unsigned short* gB = Wb + (size_t)(n0 + row) * K_DIM + seg * 32;

    // per-thread LDS store addresses (byte; low 32 bits = LDS offset)
    const unsigned sA0 = (unsigned)(uintptr_t)&As[0][row * LDSTR + seg * 32];
    const unsigned sA1 = (unsigned)(uintptr_t)&As[1][row * LDSTR + seg * 32];
    const unsigned sB0 = (unsigned)(uintptr_t)&Bs[0][row * LDSTR + seg * 32];
    const unsigned sB1 = (unsigned)(uintptr_t)&Bs[1][row * LDSTR + seg * 32];

    // per-wave fragment base pointers (lh folded in; fm/fn/ks are imm offsets)
    const unsigned short* aP0 = &As[0][(waveM * 32 + ll) * LDSTR + 8 * lh];
    const unsigned short* aP1 = &As[1][(waveM * 32 + ll) * LDSTR + 8 * lh];
    const unsigned short* bP0 = &Bs[0][(waveN * 64 + ll) * LDSTR + 8 * lh];
    const unsigned short* bP1 = &Bs[1][(waveN * 64 + ll) * LDSTR + 8 * lh];

    v8f acc[2][4] = {};

    // prologue: tile 0 -> buffer 0 (8 async b128 per thread)
#pragma unroll
    for (int j = 0; j < 4; ++j) {
        async_cp_b128(sA0 + 16 * j, gA + 8 * j);
        async_cp_b128(sB0 + 16 * j, gB + 8 * j);
    }

#pragma clang loop unroll(disable)
    for (int kt = 0; kt < NT2; ++kt) {
        const bool ib = (kt & 1) != 0;        // buffer holding the current tile
        // last iteration issues a harmless in-bounds dummy (tile 0) so the
        // wait accounting stays uniform; s_endpgm's implicit wait drains it.
        const int knext = (kt + 1 < NT2) ? (kt + 1) * KTILE : 0;
        const unsigned sAn = ib ? sA0 : sA1;  // fill the other buffer
        const unsigned sBn = ib ? sB0 : sB1;

#pragma unroll
        for (int j = 0; j < 4; ++j) {
            async_cp_b128(sAn + 16 * j, gA + knext + 8 * j);
            async_cp_b128(sBn + 16 * j, gB + knext + 8 * j);
        }

        WAIT_ASYNC(8);       // in-order completion: current tile's 8 are done
        __syncthreads();     // current tile visible to all waves

        const unsigned short* a0 = ib ? aP1 : aP0;
        const unsigned short* b0 = ib ? bP1 : bP0;

#pragma unroll
        for (int ks = 0; ks < 2; ++ks) {
            // A frag (ISA 7.12.2): lane ll = row; VGPR0-3 = K 0..7 / 8..15 (lh),
            // VGPR4-7 = K 16..23 / 24..31 within this 32-wide sub-step.
            union { v16bf v; uint4 q[2]; } af[2], bfv[4];
#pragma unroll
            for (int fm = 0; fm < 2; ++fm) {
                const unsigned short* p = a0 + fm * 16 * LDSTR + ks * 32;
                af[fm].q[0] = *(const uint4*)(p);
                af[fm].q[1] = *(const uint4*)(p + 16);
            }
            // B frag: lane ll = col; same K order as A: K 0..7 / 8..15 (lh), then K 16..23 / 24..31
#pragma unroll
            for (int fn = 0; fn < 4; ++fn) {
                const unsigned short* p = b0 + fn * 16 * LDSTR + ks * 32;
                bfv[fn].q[0] = *(const uint4*)(p);
                bfv[fn].q[1] = *(const uint4*)(p + 16);
            }
#pragma unroll
            for (int fm = 0; fm < 2; ++fm)
#pragma unroll
                for (int fn = 0; fn < 4; ++fn)
                    acc[fm][fn] = __builtin_amdgcn_wmma_f32_16x16x32_bf16(
                        /*neg_a=*/false, af[fm].v, /*neg_b=*/false, bfv[fn].v,
                        /*c_mod=*/(short)0, acc[fm][fn],
                        /*reuse_a=*/false, /*reuse_b=*/false);
        }

        __syncthreads();     // all reads done before this buffer is refilled
    }

    // epilogue: D layout — VGPR r: lanes 0-15 -> M=r, lanes 16-31 -> M=8+r, N=ll
#pragma unroll
    for (int fm = 0; fm < 2; ++fm) {
#pragma unroll
        for (int fn = 0; fn < 4; ++fn) {
            const int n = n0 + waveN * 64 + fn * 16 + ll;
#pragma unroll
            for (int r = 0; r < 8; ++r) {
                const int m = m0 + waveM * 32 + fm * 16 + 8 * lh + r;
                out[(size_t)m * N_DIM + n] = acc[fm][fn][r];
            }
        }
    }
}

extern "C" void kernel_launch(void* const* d_in, const int* in_sizes, int n_in,
                              void* d_out, int out_size, void* d_ws, size_t ws_size,
                              hipStream_t stream) {
    (void)in_sizes; (void)n_in; (void)out_size; (void)ws_size;
    const float* x        = (const float*)d_in[0];
    const int*   base_idx = (const int*)d_in[1];
    const int*   fine_idx = (const int*)d_in[2];
    const float* scale    = (const float*)d_in[3];
    const float* lut      = (const float*)d_in[4];
    float*       out      = (float*)d_out;

    unsigned short* Xb = (unsigned short*)d_ws;                 // 64 MB
    unsigned short* Wb = Xb + (size_t)M_DIM * K_DIM;            // +32 MB

    cvt_x<<<(M_DIM * (size_t)K_DIM) / (256 * 4), 256, 0, stream>>>(x, Xb);
    dequant_w<<<(N_DIM * (size_t)K_DIM) / (256 * 4), 256, 0, stream>>>(
        base_idx, fine_idx, scale, lut, Wb);

    dim3 grid(N_DIM / 128, M_DIM / 128);
    gemm_bf16<<<grid, 256, 0, stream>>>(Xb, Wb, out);
}
